// DC_layer_33062658244884
// MI455X (gfx1250) — 8-bit matrix operand order repaired and checked by reading, compiled, not run on hardware
//
#include <hip/hip_runtime.h>

typedef __attribute__((ext_vector_type(16))) int   v16i;
typedef __attribute__((ext_vector_type(8)))  float v8f;

#define NB   32
#define CIN  256
#define HIN  56
#define WIN  56
#define KOUT 256
#define HO   54
#define WO   54
#define MTOT (NB*HO*WO)

__device__ __forceinline__ unsigned char fp8_of_int(int v) {
    const unsigned long long lo = 0x4E4C4A4844403800ull;
    const unsigned long long hi = 0x5756555453525150ull;
    v &= 15;
    return (unsigned char)((((v < 8) ? lo : hi) >> ((v & 7) * 8)) & 0xFF);
}

__global__ void quant_x(const float* __restrict__ x, unsigned char* __restrict__ x8) {
    unsigned int i = blockIdx.x * blockDim.x + threadIdx.x;
    int c = i & 255;
    unsigned int p = i >> 8;
    int w = p % WIN; p /= WIN;
    int h = p % HIN;
    int n = p / HIN;
    float f = x[(((n * CIN + c) * HIN + h) * WIN) + w];
    x8[i] = fp8_of_int((int)f);
}

__global__ void quant_w(const float* __restrict__ w, unsigned char* __restrict__ w8) {
    unsigned int i = blockIdx.x * blockDim.x + threadIdx.x;
    int c  = i & 255;
    int o  = (i >> 8) & 255;
    int rs = i >> 16;
    int r = rs / 3, s = rs % 3;
    float f = w[(((o * CIN + c) * 3 + r) * 3) + s];
    w8[i] = fp8_of_int((int)f);
}

__global__ void __launch_bounds__(256)
conv_wmma(const unsigned char* __restrict__ x8,
          const unsigned char* __restrict__ w8,
          float* __restrict__ out) {
    __shared__ unsigned char bsm[2][128 * 256];

    const int tid   = threadIdx.x;
    const int lane  = tid & 31;
    const int wave  = tid >> 5;
    const int mtile = blockIdx.x * 8 + wave;
    const int ngrp  = blockIdx.y;
    const int half  = lane >> 4;
    const int l16   = lane & 15;

    const unsigned char* bpanel = w8 + (size_t)ngrp * 128 * CIN;

    auto stage = [&](int rs) {
        const unsigned char* src = bpanel + (size_t)rs * (KOUT * CIN);
        unsigned int dstBase = (unsigned int)(size_t)(&bsm[rs & 1][0]);
#pragma unroll
        for (int j = 0; j < 8; ++j) {
            const int chunk = j * 256 + tid;
            unsigned long long g = (unsigned long long)(size_t)(src + chunk * 16);
            unsigned int       l = dstBase + chunk * 16;
            asm volatile("global_load_async_to_lds_b128 %0, %1, off"
                         :: "v"(l), "v"(g) : "memory");
        }
    };

    int row = mtile * 16 + l16;
    int wo = row % WO; int t = row / WO;
    int ho = t % HO;   int n = t / HO;

    v8f acc[8] = {};

    stage(0);

    for (int rs = 0; rs < 9; ++rs) {
        if (rs < 8) {
            stage(rs + 1);
            asm volatile("s_wait_asynccnt 0x8" ::: "memory");
        } else {
            asm volatile("s_wait_asynccnt 0x0" ::: "memory");
        }
        __syncthreads();

        const unsigned char* bsrc = &bsm[rs & 1][0];
        const int r = rs / 3, s = rs % 3;
        const unsigned char* aRow =
            x8 + (((n * HIN + ho + r) * WIN) + (wo + s)) * CIN;
        __builtin_prefetch(aRow + (size_t)WIN * CIN, 0, 1);

        for (int ch = 0; ch < 2; ++ch) {
            union { unsigned int u[16]; v16i v; } A;
            const unsigned char* ab = aRow + ch * 128 + half * 16;
#pragma unroll
            for (int j = 0; j < 8; ++j) {
                uint2 q = *(const uint2*)(ab + (j >> 1) * 32 + (j & 1) * 8);
                A.u[2 * j]     = q.x;
                A.u[2 * j + 1] = q.y;
            }
#pragma unroll
            for (int nt = 0; nt < 8; ++nt) {
                const int olocal = nt * 16 + l16;
                const unsigned char* bb =
                    bsrc + olocal * CIN + ch * 128 + half * 16;
                union { uint4 q[4]; v16i v; } B;
#pragma unroll
                for (int g = 0; g < 4; ++g)
                    B.q[g] = *(const uint4*)(bb + g * 32);

                acc[nt] = __builtin_amdgcn_wmma_f32_16x16x128_fp8_fp8(
                    A.v, B.v, (short)0, acc[nt],
                     false,  false);
            }
        }
        __syncthreads();
    }

#pragma unroll
    for (int v = 0; v < 8; ++v) {
        int m = mtile * 16 + half * 8 + v;
        int wo2 = m % WO; int t2 = m / WO;
        int ho2 = t2 % HO; int n2 = t2 / HO;
        const int base = ((n2 * KOUT) * HO + ho2) * WO + wo2;
#pragma unroll
        for (int nt = 0; nt < 8; ++nt) {
            const int o = (ngrp * 8 + nt) * 16 + l16;
            out[base + o * (HO * WO)] = acc[nt][v];
        }
    }
}

extern "C" void kernel_launch(void* const* d_in, const int* in_sizes, int n_in,
                              void* d_out, int out_size, void* d_ws, size_t ws_size,
                              hipStream_t stream) {
    const float* x = (const float*)d_in[0];
    const float* w = (const float*)d_in[1];
    float* out = (float*)d_out;

    unsigned char* x8 = (unsigned char*)d_ws;
    unsigned char* w8 = x8 + (size_t)NB * HIN * WIN * CIN;

    const int xtot = NB * HIN * WIN * CIN;
    quant_x<<<xtot / 256, 256, 0, stream>>>(x, x8);

    const int wtot = 9 * KOUT * CIN;
    quant_w<<<wtot / 256, 256, 0, stream>>>(w, w8);

    dim3 grid(MTOT / 128, 2);
    conv_wmma<<<grid, 256, 0, stream>>>(x8, w8, out);
}
